// LGGP_16836271801135
// MI455X (gfx1250) — matrix operand order repaired and checked by reading, not compiled, not run on hardware
//
#include <hip/hip_runtime.h>
#include <stdint.h>

#define B_ 8
#define N_ 1024
#define M_ 1024
#define S_ 2048
#define C_ 512
#define PH_ 64
#define AH_ 1024
#define K_ 16
#define EPS_ 1e-5f

typedef __attribute__((ext_vector_type(16))) _Float16 v16h;
typedef __attribute__((ext_vector_type(8)))  float    v8f;
typedef __attribute__((ext_vector_type(4)))  uint32_t v4u;

union Frag16 { v16h v; uint32_t u[8]; v4u q[2]; _Float16 h[16]; };
union H8     { _Float16 h[8]; v4u q; };

__device__ __forceinline__ v8f wmma_f16(const Frag16& a, const Frag16& b, v8f c) {
  // D = A(16x32 f16) x B(32x16 f16) + C(16x16 f32)
  return __builtin_amdgcn_wmma_f32_16x16x32_f16(false, a.v, false, b.v,
                                                (short)0, c, false, false);
}

// Fragment loads. Layout per CDNA5 ISA 7.12.2:
//  A (16x32 f16): lane -> row M = l16; half-of-wave hw selects K block of 8;
//    halves h[0..7]  = K elements [kb+8*hw,   kb+8*hw+8)   (contiguous)
//    halves h[8..15] = K elements [kb+16+8*hw, kb+16+8*hw+8)
//  B (32x16 f16): lane -> column N = l16; same K mapping as A:
//    halves h[0..7]  = K elements [kb+8*hw,   kb+8*hw+8)
//    halves h[8..15] = K elements [kb+16+8*hw, kb+16+8*hw+8)
__device__ __forceinline__ void load_a_frag(Frag16& a, const _Float16* rowp,
                                            int kb, int hw) {
  a.q[0] = *(const v4u*)(rowp + kb + 8 * hw);
  a.q[1] = *(const v4u*)(rowp + kb + 16 + 8 * hw);
}
__device__ __forceinline__ void load_b_frag(Frag16& b, const _Float16* colp,
                                            int kb, int hw) {
  b.q[0] = *(const v4u*)(colp + kb + 8 * hw);
  b.q[1] = *(const v4u*)(colp + kb + 16 + 8 * hw);
}

// ---------------------------------------------------------------------------
// Kernel 0: weight prep. f32->f16 weights, BN fold, pos_w2 transpose,
// and fused transposed feature table featT[b][s][c] (s = concat(N, M)).
// ---------------------------------------------------------------------------
__global__ void prep_kernel(
    const float* __restrict__ feat, const float* __restrict__ featdb,
    const float* __restrict__ attn_w1, const float* __restrict__ attn_w2,
    const float* __restrict__ attn_b1, const float* __restrict__ attn_g,
    const float* __restrict__ attn_beta, const float* __restrict__ attn_m,
    const float* __restrict__ attn_v,
    const float* __restrict__ pos_b1, const float* __restrict__ pos_g,
    const float* __restrict__ pos_beta, const float* __restrict__ pos_m,
    const float* __restrict__ pos_v, const float* __restrict__ pos_w2,
    float* __restrict__ featT,
    _Float16* __restrict__ w1h, _Float16* __restrict__ w2h,
    float* __restrict__ scale1, float* __restrict__ bias1,
    float* __restrict__ ps, float* __restrict__ pb,
    float* __restrict__ w2T)
{
  const int tid = blockIdx.x * blockDim.x + threadIdx.x;
  const int stride = gridDim.x * blockDim.x;

  // featT[b][s][c], i = ((b*2048)+s)*512 + c
  for (int i = tid; i < B_ * S_ * C_; i += stride) {
    int c = i & (C_ - 1);
    int s = (i >> 9) & (S_ - 1);
    int b = i >> 20;
    float v = (s < N_) ? feat[(b * C_ + c) * N_ + s]
                       : featdb[(b * C_ + c) * M_ + (s - N_)];
    featT[i] = v;
  }
  for (int i = tid; i < AH_ * C_; i += stride) w1h[i] = (_Float16)attn_w1[i];
  for (int i = tid; i < C_ * AH_; i += stride) w2h[i] = (_Float16)attn_w2[i];
  for (int i = tid; i < AH_; i += stride) {
    float s = attn_g[i] * rsqrtf(attn_v[i] + EPS_);
    scale1[i] = s;
    bias1[i]  = (attn_b1[i] - attn_m[i]) * s + attn_beta[i];
  }
  for (int i = tid; i < PH_; i += stride) {
    float s = pos_g[i] * rsqrtf(pos_v[i] + EPS_);
    ps[i] = s;
    pb[i] = (pos_b1[i] - pos_m[i]) * s + pos_beta[i];
  }
  for (int i = tid; i < C_ * PH_; i += stride) {
    int c = i >> 6, h = i & 63;
    w2T[h * C_ + c] = pos_w2[i];
  }
}

// ---------------------------------------------------------------------------
// Kernel 1: brute-force KNN (top-16 nearest of 2048 candidates per query).
// One thread per query; register-resident sorted list; strict '<' keeps the
// earliest index on ties (matches jax.lax.top_k tie-break by lowest index).
// ---------------------------------------------------------------------------
__global__ void knn_kernel(const float* __restrict__ pcd,
                           const float* __restrict__ pcdf,
                           int* __restrict__ knn)
{
  const int b = blockIdx.y;
  const int n = blockIdx.x * blockDim.x + threadIdx.x;
  const float* Q = pcd + b * 3 * N_;
  const float qx = Q[n], qy = Q[N_ + n], qz = Q[2 * N_ + n];

  float bd[K_]; int bi[K_];
#pragma unroll
  for (int r = 0; r < K_; ++r) { bd[r] = 3.0e38f; bi[r] = 0; }

  const float* bases[2] = { pcd + b * 3 * N_, pcdf + b * 3 * M_ };
  for (int part = 0; part < 2; ++part) {
    const float* P = bases[part];
    const int off = part * N_;
    for (int s = 0; s < N_; ++s) {
      float dx = qx - P[s], dy = qy - P[N_ + s], dz = qz - P[2 * N_ + s];
      float d = dx * dx + dy * dy + dz * dz;
      if (d < bd[K_ - 1]) {
        bd[K_ - 1] = d; bi[K_ - 1] = s + off;
#pragma unroll
        for (int r = K_ - 1; r > 0; --r) {
          if (bd[r] < bd[r - 1]) {
            float td = bd[r]; bd[r] = bd[r - 1]; bd[r - 1] = td;
            int   ti = bi[r]; bi[r] = bi[r - 1]; bi[r - 1] = ti;
          }
        }
      }
    }
  }
  int* o = knn + (b * N_ + n) * K_;
#pragma unroll
  for (int r = 0; r < K_; ++r) o[r] = bi[r];
}

// ---------------------------------------------------------------------------
// Kernel 2: fused attention. One workgroup (256 threads = 8 waves) handles
// 2 queries (32 GEMM columns). LDS (dynamic, ~137KB):
//   xT   [32][512]  f16  @ 0       (GEMM1 B operand, per-column contiguous K)
//   gfpe [32][512]  f16  @ 32768   (group_feat + pos_emb, for final reduce)
//   hT   [32][1024] f16  @ 65536   (GEMM2 B operand)
//   posh [32][64]   f32  @ 131072
//   prel [3][32]    f32  @ 139264
//   sidx [32]       i32  @ 139648
// ---------------------------------------------------------------------------
#define SMEM_BYTES 139776

__global__ void __launch_bounds__(256) pt_attn_kernel(
    const float* __restrict__ pcd, const float* __restrict__ pcdf,
    const float* __restrict__ featT, const int* __restrict__ knn,
    const float* __restrict__ pos_w1, const float* __restrict__ ps,
    const float* __restrict__ pb, const float* __restrict__ w2T,
    const float* __restrict__ pos_b2,
    const _Float16* __restrict__ w1h,
    const float* __restrict__ scale1, const float* __restrict__ bias1,
    const _Float16* __restrict__ w2h, const float* __restrict__ attn_b2,
    float* __restrict__ out)
{
  extern __shared__ char smem[];
  _Float16* xT   = (_Float16*)(smem);
  _Float16* gfpe = (_Float16*)(smem + 32768);
  _Float16* hT   = (_Float16*)(smem + 65536);
  float*    posh = (float*)(smem + 131072);
  float*    prel = (float*)(smem + 139264);
  int*      sidx = (int*)(smem + 139648);

  const int tid = threadIdx.x;
  const int q0  = blockIdx.x * 2;

  // ---- Phase A1: neighbor ids + pos_rel (3 x 32)
  if (tid < 32) {
    int j = tid;
    int q = q0 + (j >> 4);
    int b = q >> 10, n = q & (N_ - 1);
    int s = knn[q * K_ + (j & 15)];
    sidx[j] = s;
    int ss = (s < N_) ? s : (s - N_);
    const float* P = (s < N_) ? (pcd + b * 3 * N_) : (pcdf + b * 3 * M_);
    const float* Q = pcd + b * 3 * N_;
    prel[j]      = Q[n]          - P[ss];
    prel[32 + j] = Q[N_ + n]     - P[N_ + ss];
    prel[64 + j] = Q[2 * N_ + n] - P[2 * N_ + ss];
  }
  __syncthreads();

  // ---- Phase A2: pos hidden = relu(bn(pos_w1 @ pos_rel))  [32 x 64]
  for (int e = tid; e < 32 * PH_; e += 256) {
    int j = e >> 6, h = e & 63;
    float lin = pos_w1[h * 3]     * prel[j]
              + pos_w1[h * 3 + 1] * prel[32 + j]
              + pos_w1[h * 3 + 2] * prel[64 + j];
    posh[j * PH_ + h] = fmaxf(lin * ps[h] + pb[h], 0.0f);
  }
  __syncthreads();

  // ---- Phase A3: pos_emb, x = feat - gf + pe, gfpe = gf + pe
  for (int e = tid; e < 32 * C_; e += 256) {
    int c = e & (C_ - 1);
    int j = e >> 9;
    int q = q0 + (j >> 4);
    int b = q >> 10, n = q & (N_ - 1);
    int s = sidx[j];
    const float* fb = featT + (size_t)b * S_ * C_;
    float gf = fb[(size_t)s * C_ + c];
    float ft = fb[(size_t)n * C_ + c];
    float pe = pos_b2[c];
    const float* phr = posh + j * PH_;
#pragma unroll 8
    for (int h = 0; h < PH_; ++h) pe += w2T[h * C_ + c] * phr[h];
    xT[j * C_ + c]   = (_Float16)(ft - gf + pe);
    gfpe[j * C_ + c] = (_Float16)(gf + pe);
  }
  __syncthreads();

  const int lane = tid & 31;
  const int wave = tid >> 5;
  const int l16  = lane & 15;
  const int hw   = lane >> 4;   // which 16-lane half of the wave

  // ---- Phase B: GEMM1  h[1024][32] = relu(scale*(w1h @ x) + bias) -> hT f16
  for (int i = 0; i < 8; ++i) {
    const int mBase = (wave * 8 + i) * 16;
    v8f acc0 = {}; v8f acc1 = {};
    const _Float16* Arow = w1h + (size_t)(mBase + l16) * C_;
    const _Float16* b0p = xT + (size_t)l16 * C_;
    const _Float16* b1p = xT + (size_t)(16 + l16) * C_;
    for (int kb = 0; kb < C_; kb += 32) {
      Frag16 a, b0, b1;
      load_a_frag(a, Arow, kb, hw);
      load_b_frag(b0, b0p, kb, hw);
      load_b_frag(b1, b1p, kb, hw);
      acc0 = wmma_f16(a, b0, acc0);
      acc1 = wmma_f16(a, b1, acc1);
    }
    const int mOff = mBase + 8 * hw;   // rows handled by this lane half
    float sc[8], bs[8];
#pragma unroll
    for (int r = 0; r < 8; ++r) { sc[r] = scale1[mOff + r]; bs[r] = bias1[mOff + r]; }
    H8 h0, h1;
#pragma unroll
    for (int r = 0; r < 8; ++r) {
      h0.h[r] = (_Float16)fmaxf(acc0[r] * sc[r] + bs[r], 0.0f);
      h1.h[r] = (_Float16)fmaxf(acc1[r] * sc[r] + bs[r], 0.0f);
    }
    *(v4u*)(hT + (size_t)l16 * AH_ + mOff)        = h0.q;
    *(v4u*)(hT + (size_t)(16 + l16) * AH_ + mOff) = h1.q;
  }
  __syncthreads();

  // ---- Phase C: GEMM2 w = w2h @ h + b2, then softmax over neighbors and
  //      weighted reduce, all in-register via 16-lane xor shuffles.
  for (int i = 0; i < 4; ++i) {
    const int mBase = (wave * 4 + i) * 16;
    v8f acc0 = {}; v8f acc1 = {};
    const _Float16* Arow = w2h + (size_t)(mBase + l16) * AH_;
    const _Float16* b0p = hT + (size_t)l16 * AH_;
    const _Float16* b1p = hT + (size_t)(16 + l16) * AH_;
    for (int kb = 0; kb < AH_; kb += 32) {
      Frag16 a, b0, b1;
      load_a_frag(a, Arow, kb, hw);
      load_b_frag(b0, b0p, kb, hw);
      load_b_frag(b1, b1p, kb, hw);
      acc0 = wmma_f16(a, b0, acc0);
      acc1 = wmma_f16(a, b1, acc1);
    }
    const int mOff = mBase + 8 * hw;
    float bb[8];
#pragma unroll
    for (int r = 0; r < 8; ++r) bb[r] = attn_b2[mOff + r];

#pragma unroll
    for (int jT = 0; jT < 2; ++jT) {
      v8f accv = jT ? acc1 : acc0;
      float w[8];
#pragma unroll
      for (int r = 0; r < 8; ++r) w[r] = accv[r] + bb[r];
      // softmax across the 16 neighbor columns (lanes within each half;
      // xor masks < 16 never cross the 16-lane boundary)
      float mx[8];
#pragma unroll
      for (int r = 0; r < 8; ++r) mx[r] = w[r];
#pragma unroll
      for (int d = 1; d < 16; d <<= 1) {
#pragma unroll
        for (int r = 0; r < 8; ++r) mx[r] = fmaxf(mx[r], __shfl_xor(mx[r], d, 32));
      }
      float ex[8], sm[8];
#pragma unroll
      for (int r = 0; r < 8; ++r) { ex[r] = __expf(w[r] - mx[r]); sm[r] = ex[r]; }
#pragma unroll
      for (int d = 1; d < 16; d <<= 1) {
#pragma unroll
        for (int r = 0; r < 8; ++r) sm[r] += __shfl_xor(sm[r], d, 32);
      }
      H8 g;
      const int j = jT * 16 + l16;
      g.q = *(const v4u*)(gfpe + (size_t)j * C_ + mOff);
      float red[8];
#pragma unroll
      for (int r = 0; r < 8; ++r) red[r] = (ex[r] / sm[r]) * (float)g.h[r];
#pragma unroll
      for (int d = 1; d < 16; d <<= 1) {
#pragma unroll
        for (int r = 0; r < 8; ++r) red[r] += __shfl_xor(red[r], d, 32);
      }
      if (l16 == 0) {
        int q = q0 + jT;
        int b = q >> 10, n = q & (N_ - 1);
        float* o = out + ((size_t)b * C_ + mOff) * N_ + n;
#pragma unroll
        for (int r = 0; r < 8; ++r) o[(size_t)r * N_] = red[r];
      }
    }
  }
}

// ---------------------------------------------------------------------------
extern "C" void kernel_launch(void* const* d_in, const int* in_sizes, int n_in,
                              void* d_out, int out_size, void* d_ws, size_t ws_size,
                              hipStream_t stream) {
  const float* pcd       = (const float*)d_in[0];
  const float* feat      = (const float*)d_in[1];
  const float* pcdf      = (const float*)d_in[2];
  const float* featdb    = (const float*)d_in[3];
  const float* pos_w1    = (const float*)d_in[4];
  const float* pos_b1    = (const float*)d_in[5];
  const float* pos_g     = (const float*)d_in[6];
  const float* pos_beta  = (const float*)d_in[7];
  const float* pos_m     = (const float*)d_in[8];
  const float* pos_v     = (const float*)d_in[9];
  const float* pos_w2    = (const float*)d_in[10];
  const float* pos_b2    = (const float*)d_in[11];
  const float* attn_w1   = (const float*)d_in[12];
  const float* attn_b1   = (const float*)d_in[13];
  const float* attn_g    = (const float*)d_in[14];
  const float* attn_beta = (const float*)d_in[15];
  const float* attn_m    = (const float*)d_in[16];
  const float* attn_v    = (const float*)d_in[17];
  const float* attn_w2   = (const float*)d_in[18];
  const float* attn_b2   = (const float*)d_in[19];
  float* out = (float*)d_out;

  char* ws = (char*)d_ws;
  float*    featT  = (float*)(ws + 0);           // 32 MiB
  int*      knn    = (int*)(ws + 33554432);      // 512 KiB
  _Float16* w1h    = (_Float16*)(ws + 34078720); // 1 MiB
  _Float16* w2h    = (_Float16*)(ws + 35127296); // 1 MiB
  float*    scale1 = (float*)(ws + 36175872);
  float*    bias1  = (float*)(ws + 36179968);
  float*    ps     = (float*)(ws + 36184064);
  float*    pb     = (float*)(ws + 36184320);
  float*    w2T    = (float*)(ws + 36184576);    // 128 KiB, end ~34.6 MiB

  prep_kernel<<<dim3(2048), dim3(256), 0, stream>>>(
      feat, featdb, attn_w1, attn_w2, attn_b1, attn_g, attn_beta, attn_m,
      attn_v, pos_b1, pos_g, pos_beta, pos_m, pos_v, pos_w2,
      featT, w1h, w2h, scale1, bias1, ps, pb, w2T);

  knn_kernel<<<dim3(8, 8), dim3(128), 0, stream>>>(pcd, pcdf, knn);

  pt_attn_kernel<<<dim3((B_ * N_) / 2), dim3(256), SMEM_BYTES, stream>>>(
      pcd, pcdf, featT, knn, pos_w1, ps, pb, w2T, pos_b2,
      w1h, scale1, bias1, w2h, attn_b2, out);
}
